// EDeeperGCN_70909910057020
// MI455X (gfx1250) — matrix operand order repaired and checked by reading, not compiled, not run on hardware
//
#include <hip/hip_runtime.h>
#include <hip/hip_bf16.h>

typedef __attribute__((ext_vector_type(16))) __bf16 v16bf;
typedef __attribute__((ext_vector_type(8)))  float  v8f;

#define N_NODES 10000
#define N_EDGES 640000
#define HIDDEN  128
#define OUTDIM  10

union Frag {
    v16bf bf;
    int4  i[2];
};

__device__ __forceinline__ unsigned short f2bf_bits(float f) {
    unsigned u = __builtin_bit_cast(unsigned, f);
    unsigned r = (u + 0x7FFFu + ((u >> 16) & 1u)) >> 16;   // round to nearest even
    return (unsigned short)r;
}
__device__ __forceinline__ __bf16 f2bf(float f) {
    unsigned short s = f2bf_bits(f);
    return __builtin_bit_cast(__bf16, s);
}

// ---------------- prep kernel 1: x (f32) -> bf16 ----------------
__global__ void cvt_x_kernel(const float* __restrict__ x, __bf16* __restrict__ xb, int n) {
    int i = blockIdx.x * blockDim.x + threadIdx.x;
    if (i < n) xb[i] = f2bf(x[i]);
}

// ---------------- prep kernel 2: pack W1 into WMMA-B fragment layout ----------
// lane L holds column n = L%16; K-run = kstep*32 + (L>=16 ? 16 : 0) + [0..15]
// dword j holds packed pair (K = run+2j, run+2j+1)
// Storage: w1p[ ((nt*8 + ks)*32 + lane)*8 + j ]
__global__ void pack_w1_kernel(const float* __restrict__ W1, unsigned* __restrict__ w1p) {
    int idx = blockIdx.x * blockDim.x + threadIdx.x;   // 8*8*32*8 = 16384
    if (idx >= 8 * 8 * 32 * 8) return;
    int j    = idx & 7;
    int lane = (idx >> 3) & 31;
    int ks   = (idx >> 8) & 7;
    int nt   = idx >> 11;
    int n = nt * 16 + (lane & 15);
    int k = ks * 32 + ((lane >> 4) ? 16 : 0) + 2 * j;
    unsigned lo = f2bf_bits(W1[k * HIDDEN + n]);
    unsigned hi = f2bf_bits(W1[(k + 1) * HIDDEN + n]);
    w1p[idx] = (hi << 16) | lo;
}

// ---------------- prep kernel 3: pack W2 (N padded 10 -> 16) ----------------
// Storage: w2p[ (ks*32 + lane)*8 + j ], ks in 0..3
__global__ void pack_w2_kernel(const float* __restrict__ W2, unsigned* __restrict__ w2p) {
    int idx = blockIdx.x * blockDim.x + threadIdx.x;   // 4*32*8 = 1024
    if (idx >= 4 * 32 * 8) return;
    int j    = idx & 7;
    int lane = (idx >> 3) & 31;
    int ks   = idx >> 8;
    int n = lane & 15;
    int k = ks * 32 + ((lane >> 4) ? 16 : 0) + 2 * j;
    float v0 = (n < OUTDIM) ? W2[k * OUTDIM + n] : 0.f;
    float v1 = (n < OUTDIM) ? W2[(k + 1) * OUTDIM + n] : 0.f;
    w2p[idx] = ((unsigned)f2bf_bits(v1) << 16) | f2bf_bits(v0);
}

// ---------------- main fused kernel: gather + MLP2 via WMMA ----------------
// block = 256 threads = 8 waves; each wave owns 32 edges (two M-tiles),
// so every W1 B-fragment loaded from LDS feeds TWO WMMAs. B is manually
// double-buffered so a DS load is always in flight behind the WMMA pair.
__global__ void __launch_bounds__(256)
edge_mlp_kernel(const int* __restrict__ ei,
                const __bf16* __restrict__ xb,
                const unsigned* __restrict__ w1p,
                const unsigned* __restrict__ w2p,
                const float* __restrict__ b1,
                const float* __restrict__ b2,
                float* __restrict__ out) {
    __shared__ unsigned w1s[8 * 8 * 32 * 8];    // 64 KB: packed W1 fragments
    __shared__ __bf16  hbuf[8 * 32 * HIDDEN];   // 64 KB: per-wave 32x128 hidden tile

    const int tid  = threadIdx.x;
    const int lane = tid & 31;
    const int wid  = tid >> 5;
    const int tile = blockIdx.x * 8 + wid;      // 0 .. 19999
    const int e0   = tile * 32;
    const int row  = lane & 15;                 // M index within a 16-row tile
    const int half = lane >> 4;                 // 0 / 1
    const int off  = half * 16;                 // A-fragment K sub-offset (elements)

    // ---- stage packed W1 into LDS (64 KB, 256 threads x 16 x b128)
#pragma unroll
    for (int i = 0; i < 16; ++i) {
        ((int4*)w1s)[tid + i * 256] = ((const int4*)w1p)[tid + i * 256];
    }

    // two edges per lane (M-tile 0: rows 0..15, M-tile 1: rows 16..31)
    const int eA = e0 + row;
    const int eB = e0 + 16 + row;
    const int srcA = ei[eA], dstA = ei[N_EDGES + eA];
    const int srcB = ei[eB], dstB = ei[N_EDGES + eB];
    const __bf16* srcAp = xb + (size_t)srcA * HIDDEN;
    const __bf16* dstAp = xb + (size_t)dstA * HIDDEN;
    const __bf16* srcBp = xb + (size_t)srcB * HIDDEN;
    const __bf16* dstBp = xb + (size_t)dstB * HIDDEN;

    // ---- layer 1 accumulators, init with bias b1 (depends only on column)
    v8f c0[8], c1[8];
#pragma unroll
    for (int nt = 0; nt < 8; ++nt) {
        float b = b1[nt * 16 + row];
        c0[nt] = (v8f){b, b, b, b, b, b, b, b};
        c1[nt] = c0[nt];
    }

    __syncthreads();   // W1 staged (full EXEC: before any divergence)

    const unsigned* w1base = w1s + (lane << 3);

    // ---- layer 1: [32,256] @ [256,128]; K-steps of 32; A prefetched one step
    //      ahead, B double-buffered from LDS.
    Frag a0, a1;
    {
        const __bf16* rp0 = srcAp + off;
        const __bf16* rp1 = srcBp + off;
        a0.i[0] = *(const int4*)(rp0);  a0.i[1] = *(const int4*)(rp0 + 8);
        a1.i[0] = *(const int4*)(rp1);  a1.i[1] = *(const int4*)(rp1 + 8);
    }
#pragma unroll
    for (int ks = 0; ks < 8; ++ks) {
        Frag a0n, a1n;
        if (ks < 7) {
            const int kn = ks + 1;
            const int cb = (kn & 3) * 32 + off;
            const __bf16* rp0 = ((kn < 4) ? srcAp : dstAp) + cb;
            const __bf16* rp1 = ((kn < 4) ? srcBp : dstBp) + cb;
            a0n.i[0] = *(const int4*)(rp0);  a0n.i[1] = *(const int4*)(rp0 + 8);
            a1n.i[0] = *(const int4*)(rp1);  a1n.i[1] = *(const int4*)(rp1 + 8);
        }
        // double-buffered B fragments from LDS
        Frag bc, bn;
        {
            const unsigned* wp = w1base + ((0 * 8 + ks) << 8);   // (nt*8+ks)*32*8
            bc.i[0] = *(const int4*)(wp);
            bc.i[1] = *(const int4*)(wp + 4);
        }
#pragma unroll
        for (int nt = 0; nt < 8; ++nt) {
            if (nt < 7) {
                const unsigned* wp = w1base + (((nt + 1) * 8 + ks) << 8);
                bn.i[0] = *(const int4*)(wp);
                bn.i[1] = *(const int4*)(wp + 4);
            }
            c0[nt] = __builtin_amdgcn_wmma_f32_16x16x32_bf16(
                false, a0.bf, false, bc.bf, (short)0, c0[nt], false, false);
            c1[nt] = __builtin_amdgcn_wmma_f32_16x16x32_bf16(
                false, a1.bf, false, bc.bf, (short)0, c1[nt], false, false);
            if (nt < 7) bc = bn;
        }
        if (ks < 7) { a0 = a0n; a1 = a1n; }
    }

    // ---- ReLU + stage 32x128 hidden tile to LDS as bf16 (D layout -> row-major)
    __bf16* hb = hbuf + wid * (32 * HIDDEN);
#pragma unroll
    for (int nt = 0; nt < 8; ++nt) {
        const int n = nt * 16 + row;
#pragma unroll
        for (int r = 0; r < 8; ++r) {
            const int m = r + half * 8;
            hb[m * HIDDEN + n]        = f2bf(fmaxf(c0[nt][r], 0.f));
            hb[(m + 16) * HIDDEN + n] = f2bf(fmaxf(c1[nt][r], 0.f));
        }
    }
    // same-wave DS ordering + explicit wait: all lanes' LDS stores visible
    asm volatile("s_wait_dscnt 0" ::: "memory");

    // ---- layer 2: [32,128] @ [128,16]; each B fragment feeds two WMMAs
    float b2v = (row < OUTDIM) ? b2[row] : 0.f;
    v8f d0 = (v8f){b2v, b2v, b2v, b2v, b2v, b2v, b2v, b2v};
    v8f d1 = d0;
#pragma unroll
    for (int ks = 0; ks < 4; ++ks) {
        const __bf16* rp0 = hb + row * HIDDEN + ks * 32 + off;
        const __bf16* rp1 = rp0 + 16 * HIDDEN;
        Frag a2, a3, b;
        a2.i[0] = *(const int4*)(rp0);  a2.i[1] = *(const int4*)(rp0 + 8);
        a3.i[0] = *(const int4*)(rp1);  a3.i[1] = *(const int4*)(rp1 + 8);
        const unsigned* wp = w2p + (((ks * 32) + lane) << 3);
        b.i[0] = *(const int4*)(wp);
        b.i[1] = *(const int4*)(wp + 4);
        d0 = __builtin_amdgcn_wmma_f32_16x16x32_bf16(
            false, a2.bf, false, b.bf, (short)0, d0, false, false);
        d1 = __builtin_amdgcn_wmma_f32_16x16x32_bf16(
            false, a3.bf, false, b.bf, (short)0, d1, false, false);
    }

    // ---- store output [32 edges, 10] (divergence only after all WMMAs)
    if (row < OUTDIM) {
#pragma unroll
        for (int r = 0; r < 8; ++r) {
            const int m = r + half * 8;
            out[(size_t)(e0 + m) * OUTDIM + row]      = d0[r];
            out[(size_t)(e0 + 16 + m) * OUTDIM + row] = d1[r];
        }
    }
}

extern "C" void kernel_launch(void* const* d_in, const int* in_sizes, int n_in,
                              void* d_out, int out_size, void* d_ws, size_t ws_size,
                              hipStream_t stream) {
    const float* x  = (const float*)d_in[0];
    const int*   ei = (const int*)d_in[1];
    const float* W1 = (const float*)d_in[2];
    const float* b1 = (const float*)d_in[3];
    const float* W2 = (const float*)d_in[4];
    const float* b2 = (const float*)d_in[5];
    float* out = (float*)d_out;

    // workspace layout
    char* ws = (char*)d_ws;
    __bf16*   xb  = (__bf16*)ws;                                    // 2,560,000 B
    unsigned* w1p = (unsigned*)(ws + (size_t)N_NODES * HIDDEN * 2); // 65,536 B
    unsigned* w2p = (unsigned*)(ws + (size_t)N_NODES * HIDDEN * 2 + 16384 * 4); // 4,096 B

    // prep: convert x to bf16, pack weights into WMMA B-fragment layout
    cvt_x_kernel<<<(N_NODES * HIDDEN + 255) / 256, 256, 0, stream>>>(x, xb, N_NODES * HIDDEN);
    pack_w1_kernel<<<(16384 + 255) / 256, 256, 0, stream>>>(W1, w1p);
    pack_w2_kernel<<<(1024 + 255) / 256, 256, 0, stream>>>(W2, w2p);

    // main: 20000 wave-tiles of 32 edges, 8 waves/block -> 2500 blocks
    edge_mlp_kernel<<<N_EDGES / 32 / 8, 256, 0, stream>>>(ei, xb, w1p, w2p, b1, b2, out);
}
